// MOTASG_KO_Reg_7713761264093
// MI455X (gfx1250) — matrix operand order repaired and checked by reading, not compiled, not run on hardware
//
#include <hip/hip_runtime.h>

#define LEAKY 0.3f

// Model constants (from reference): NE=1024, B=16, K=64, TX=768, OM=511, D=512, N=16384
#define NROW   16384
#define NE_    1024
#define TX_    768
#define DPAD   512
#define BK_    1024   // B*K rows for readout
#define BIGF   ((size_t)NROW * DPAD)          // floats per big slot (8,388,608)
#define CNEF   ((size_t)NE_ * DPAD)           // crossNE floats (524,288)

typedef __attribute__((ext_vector_type(16))) __bf16 bf16x16;
typedef __attribute__((ext_vector_type(8)))  float  f32x8;
typedef __attribute__((ext_vector_type(4)))  float  f32x4;

// ---------------------------------------------------------------------------
// WMMA GEMM: C[M x N] (ldc) = op( A[M x K](lda,f32) @ Bt^T ) where Bt is the
// weight stored transposed as [N x Kpad] bf16 row-major (ldb = Kpad).
// Wave tile 16x64 (4 x 16x16 accumulators), block = 8 waves = 128 rows.
// act: 0=none 1=leaky_relu(0.3) 2=tanh ; addC: 1 -> C += result.
// Requires: M %128==0, N %64==0, K %32==0, lda/ldb/ldc %4==0.
// ---------------------------------------------------------------------------
__global__ __launch_bounds__(256)
void k_gemm(const float* __restrict__ A, int lda,
            const __bf16* __restrict__ Bt, int ldb,
            const float* __restrict__ bias,
            float* __restrict__ C, int ldc,
            int K, int act, int addC)
{
    const int lane = threadIdx.x & 31;
    const int wave = threadIdx.x >> 5;
    const int r    = lane & 15;
    const int half = lane >> 4;
    const int m0   = blockIdx.y * 128 + wave * 16;
    const int n0   = blockIdx.x * 64;

    f32x8 acc[4] = {};
    const float* arow = A + (size_t)(m0 + r) * lda;

    for (int k = 0; k < K; k += 32) {
        // A fragment: row (m0+r); K elements {k+half*16 .. +15}
        bf16x16 a;
        const float* ap = arow + k + half * 16;
        f32x4 a0 = *(const f32x4*)(ap);
        f32x4 a1 = *(const f32x4*)(ap + 4);
        f32x4 a2 = *(const f32x4*)(ap + 8);
        f32x4 a3 = *(const f32x4*)(ap + 12);
#pragma unroll
        for (int i = 0; i < 4; ++i) {
            a[i]      = (__bf16)a0[i];
            a[4 + i]  = (__bf16)a1[i];
            a[8 + i]  = (__bf16)a2[i];
            a[12 + i] = (__bf16)a3[i];
        }
#pragma unroll
        for (int j = 0; j < 4; ++j) {
            // B fragment: column (n0+j*16+r), rows k+half*16 .. k+half*16+15 (contiguous in Bt)
            const __bf16* bp = Bt + (size_t)(n0 + j * 16 + r) * ldb + k + half * 16;
            bf16x16 b = *(const bf16x16*)bp;
            acc[j] = __builtin_amdgcn_wmma_f32_16x16x32_bf16(
                false, a, false, b, (short)0, acc[j], false, false);
        }
    }

#pragma unroll
    for (int j = 0; j < 4; ++j) {
        const int col = n0 + j * 16 + r;
        const float bv = bias ? bias[col] : 0.0f;
#pragma unroll
        for (int i = 0; i < 8; ++i) {
            const int row = m0 + i + half * 8;
            float v = acc[j][i] + bv;
            if (act == 1)      v = (v > 0.0f) ? v : v * LEAKY;
            else if (act == 2) v = tanhf(v);
            float* p = C + (size_t)row * ldc + col;
            if (addC) *p += v; else *p = v;
        }
    }
}

// Stage weight W[K x Nn] (row-major, leading dim ldw) into Bt[Npad x Kpad] bf16,
// transposed + zero padded.
__global__ void k_stage_bt(const float* __restrict__ W, int ldw, int K, int Nn,
                           __bf16* __restrict__ Bt, int Kpad, int total)
{
    int idx = blockIdx.x * 256 + threadIdx.x;
    if (idx >= total) return;
    int n = idx / Kpad, k = idx % Kpad;
    float v = (k < K && n < Nn) ? W[(size_t)k * ldw + n] : 0.0f;
    Bt[idx] = (__bf16)v;
}

// Pad a 511-wide bias to 512 (pad = 0)
__global__ void k_pad_bias(float* __restrict__ out, const float* __restrict__ src)
{
    int c = blockIdx.x * 256 + threadIdx.x;
    if (c < 512) out[c] = (c < 511) ? src[c] : 0.0f;
}

// Pad N x 511 activation to N x 512 (pad column = 0)
__global__ void k_pad511(float* __restrict__ out, const float* __restrict__ src, int total)
{
    int idx = blockIdx.x * 256 + threadIdx.x;
    if (idx >= total) return;
    int n = idx >> 9, c = idx & 511;
    out[idx] = (c < 511) ? src[(size_t)n * 511 + c] : 0.0f;
}

// Set KO indicator column (col 511) to 1.0 at rows in ko_mask
__global__ void k_ko(float* __restrict__ base, const int* __restrict__ km, int n)
{
    int i = blockIdx.x * 256 + threadIdx.x;
    if (i < n) base[(size_t)km[i] * 512 + 511] = 1.0f;
}

// cross[N x 512] += crossNE[(n % 1024) x 512]   (1024*512 = 524288, power of two)
__global__ void k_addtile(float* __restrict__ C, const float* __restrict__ T, int total)
{
    int idx = blockIdx.x * 256 + threadIdx.x;
    if (idx < total) C[idx] += T[idx & (int)(CNEF - 1)];
}

// Edge scatter: acc[dst] += m[src] over 512 columns (atomic f32)
__global__ void k_edge(const int* __restrict__ ei, int E,
                       const float* __restrict__ m, float* __restrict__ acc)
{
    int e = blockIdx.x;
    int src = ei[e], dst = ei[E + e];
    const float* ms = m + (size_t)src * 512;
    float* ad = acc + (size_t)dst * 512;
    atomicAdd(&ad[threadIdx.x],        ms[threadIdx.x]);
    atomicAdd(&ad[threadIdx.x + 256],  ms[threadIdx.x + 256]);
}

// out = leaky(acc + bias[col]) (+ addSrc)
__global__ void k_finz(float* __restrict__ out, const float* __restrict__ acc,
                       const float* __restrict__ bias, const float* __restrict__ addSrc,
                       int total)
{
    int idx = blockIdx.x * 256 + threadIdx.x;
    if (idx >= total) return;
    float v = acc[idx] + bias[idx & 511];
    v = (v > 0.0f) ? v : v * LEAKY;
    if (addSrc) v += addSrc[idx];
    out[idx] = v;
}

// zk[row] = z2[ko_mask[row]]  (row in [0,1024), 512 cols)
__global__ void k_gather(float* __restrict__ zk, const float* __restrict__ z2,
                         const int* __restrict__ km)
{
    int row = blockIdx.x;
    int g = km[row];
    zk[(size_t)row * 512 + threadIdx.x]       = z2[(size_t)g * 512 + threadIdx.x];
    zk[(size_t)row * 512 + threadIdx.x + 256] = z2[(size_t)g * 512 + threadIdx.x + 256];
}

// Per row (1024 rows): scores[i] = g[i].W2 + b2 ; t[i] = zk[i].regW
__global__ __launch_bounds__(256)
void k_rowdot(const float* __restrict__ g, const float* __restrict__ zk,
              const float* __restrict__ w2, const float* __restrict__ b2,
              const float* __restrict__ regW,
              float* __restrict__ scores, float* __restrict__ t)
{
    int row  = blockIdx.x * 8 + (threadIdx.x >> 5);
    int lane = threadIdx.x & 31;
    const float* gr = g  + (size_t)row * 512;
    const float* zr = zk + (size_t)row * 512;
    float s = 0.0f, tt = 0.0f;
    for (int c = lane; c < 512; c += 32) {
        s  += gr[c] * w2[c];
        tt += zr[c] * regW[c];
    }
#pragma unroll
    for (int off = 16; off > 0; off >>= 1) {
        s  += __shfl_xor(s,  off, 32);
        tt += __shfl_xor(tt, off, 32);
    }
    if (lane == 0) { scores[row] = s + b2[0]; t[row] = tt; }
}

// Per batch b (16 blocks, 64 threads): out[b] = sum_k softmax(scores)_k * t_k + reg_b
__global__ void k_softmax_out(const float* __restrict__ scores, const float* __restrict__ t,
                              const float* __restrict__ reg_b, float* __restrict__ out)
{
    __shared__ float sm[64];
    int b = blockIdx.x, k = threadIdx.x;
    float s = scores[b * 64 + k];
    sm[k] = s; __syncthreads();
    for (int off = 32; off > 0; off >>= 1) {
        if (k < off) sm[k] = fmaxf(sm[k], sm[k + off]);
        __syncthreads();
    }
    float mx = sm[0]; __syncthreads();
    float e = expf(s - mx);
    sm[k] = e; __syncthreads();
    for (int off = 32; off > 0; off >>= 1) {
        if (k < off) sm[k] += sm[k + off];
        __syncthreads();
    }
    float denom = sm[0]; __syncthreads();
    float val = (e / denom) * t[b * 64 + k];
    sm[k] = val; __syncthreads();
    for (int off = 32; off > 0; off >>= 1) {
        if (k < off) sm[k] += sm[k + off];
        __syncthreads();
    }
    if (k == 0) out[b] = sm[0] + reg_b[0];
}

// ---------------------------------------------------------------------------

static inline void gemm(hipStream_t st, const float* A, int lda, const __bf16* Bt, int ldb,
                        const float* bias, float* C, int ldc,
                        int M, int Nn, int K, int act, int addC)
{
    dim3 g(Nn / 64, M / 128);
    k_gemm<<<g, 256, 0, st>>>(A, lda, Bt, ldb, bias, C, ldc, K, act, addC);
}

extern "C" void kernel_launch(void* const* d_in, const int* in_sizes, int n_in,
                              void* d_out, int out_size, void* d_ws, size_t ws_size,
                              hipStream_t stream)
{
    // Inputs (setup_inputs dict order)
    const float* x        = (const float*)d_in[0];
    const float* pre_x    = (const float*)d_in[1];
    const int*   edge     = (const int*)d_in[2];   // 2 x 262144
    const int*   iedge    = (const int*)d_in[3];   // 2 x 131072
    const float* name_emb = (const float*)d_in[6];
    const float* desc_emb = (const float*)d_in[7];
    const int*   ko_mask  = (const int*)d_in[10];  // 1024
    const float* name_W   = (const float*)d_in[11];
    const float* name_b   = (const float*)d_in[12];
    const float* desc_W   = (const float*)d_in[13];
    const float* desc_b   = (const float*)d_in[14];
    const float* omic_W   = (const float*)d_in[15];
    const float* omic_b   = (const float*)d_in[16];
    const float* fus_W    = (const float*)d_in[17]; // 2047 x 511
    const float* fus_b    = (const float*)d_in[18];
    const float* pre_W    = (const float*)d_in[19]; // 512 x 512
    const float* pre_b    = (const float*)d_in[20];
    const float* ienc_W   = (const float*)d_in[21];
    const float* ienc_b   = (const float*)d_in[22];
    const float* enc_W    = (const float*)d_in[23];
    const float* enc_b    = (const float*)d_in[24];
    const float* gate_W1  = (const float*)d_in[25];
    const float* gate_b1  = (const float*)d_in[26];
    const float* gate_W2  = (const float*)d_in[27];
    const float* gate_b2  = (const float*)d_in[28];
    const float* reg_W    = (const float*)d_in[29];
    const float* reg_b    = (const float*)d_in[30];
    float* out = (float*)d_out;

    // Workspace layout: 5 big N x 512 f32 slots + bf16 weight region + padded biases
    float* S0 = (float*)d_ws;          // xpad (=x_c), later zk
    float* S1 = S0 + BIGF;             // omic -> prepad -> z2
    float* S2 = S1 + BIGF;             // cross(_c) -> z
    float* S3 = S2 + BIGF;             // nameL -> m -> m2 -> g
    float* S4 = S3 + BIGF;             // descL+crossNE -> acc -> acc2 -> scores/t
    __bf16* WT = (__bf16*)(S4 + BIGF);
    __bf16* nameWt = WT;                      // 768x768
    __bf16* descWt = nameWt + 768 * 768;      // 768x768
    __bf16* WnT    = descWt + 768 * 768;      // 512x768
    __bf16* WdT    = WnT + 512 * 768;         // 512x768
    __bf16* WoT    = WdT + 512 * 768;         // 512x512
    __bf16* omicWt = WoT + 512 * 512;
    __bf16* iencWt = omicWt + 512 * 512;
    __bf16* encWt  = iencWt + 512 * 512;
    __bf16* preWt  = encWt + 512 * 512;
    __bf16* g1Wt   = preWt + 512 * 512;
    float* biasOm  = (float*)(g1Wt + 512 * 512); // 512
    float* biasFus = biasOm + 512;               // 512
    size_t needed = (size_t)((char*)(biasFus + 512) - (char*)d_ws);
    if (ws_size < needed) return;

    float* nameL   = S3;               // 1024 x 768
    float* descL   = S4;               // 1024 x 768
    float* crossNE = S4 + 1048576;     // 1024 x 512 (past descL)
    const size_t TOT = BIGF;           // 8,388,608 elementwise size
    const int EWB = (int)(TOT / 256);  // 32768 blocks

    // ---- Stage weights (transpose + pad + bf16) ----
    k_stage_bt<<<(768*768)/256, 256, 0, stream>>>(name_W, 768, 768, 768, nameWt, 768, 768*768);
    k_stage_bt<<<(768*768)/256, 256, 0, stream>>>(desc_W, 768, 768, 768, descWt, 768, 768*768);
    k_stage_bt<<<(512*768)/256, 256, 0, stream>>>(fus_W,            511, 768, 511, WnT, 768, 512*768);
    k_stage_bt<<<(512*768)/256, 256, 0, stream>>>(fus_W + 768*511,  511, 768, 511, WdT, 768, 512*768);
    k_stage_bt<<<(512*512)/256, 256, 0, stream>>>(fus_W + 1536*511, 511, 511, 511, WoT, 512, 512*512);
    k_stage_bt<<<(512*512)/256, 256, 0, stream>>>(omic_W, 511, 511, 511, omicWt, 512, 512*512);
    k_stage_bt<<<(512*512)/256, 256, 0, stream>>>(ienc_W, 512, 512, 512, iencWt, 512, 512*512);
    k_stage_bt<<<(512*512)/256, 256, 0, stream>>>(enc_W,  512, 512, 512, encWt,  512, 512*512);
    k_stage_bt<<<(512*512)/256, 256, 0, stream>>>(pre_W,  512, 512, 512, preWt,  512, 512*512);
    k_stage_bt<<<(512*512)/256, 256, 0, stream>>>(gate_W1,512, 512, 512, g1Wt,   512, 512*512);
    k_pad_bias<<<2, 256, 0, stream>>>(biasOm,  omic_b);
    k_pad_bias<<<2, 256, 0, stream>>>(biasFus, fus_b);

    // ---- x_c = [x | ko] padded to N x 512 ----
    k_pad511<<<EWB, 256, 0, stream>>>(S0, x, (int)TOT);
    k_ko<<<4, 256, 0, stream>>>(S0, ko_mask, 1024);

    // ---- name/desc branches (NE rows only: tile(A)@W == tile(A@W)) ----
    gemm(stream, name_emb, 768, nameWt, 768, name_b, nameL, 768, NE_, 768, 768, 1, 0);
    gemm(stream, desc_emb, 768, descWt, 768, desc_b, descL, 768, NE_, 768, 768, 1, 0);
    gemm(stream, nameL, 768, WnT, 768, nullptr, crossNE, 512, NE_, 512, 768, 0, 0);
    gemm(stream, descL, 768, WdT, 768, nullptr, crossNE, 512, NE_, 512, 768, 0, 1);

    // ---- omic = leaky(x @ omic_W + b) ----
    gemm(stream, S0, 512, omicWt, 512, biasOm, S1, 512, NROW, 512, 512, 1, 0);

    // ---- cross_x = omic @ Wo + fus_b + tile(crossNE); then ko column -> cross_c ----
    gemm(stream, S1, 512, WoT, 512, biasFus, S2, 512, NROW, 512, 512, 0, 0);
    k_addtile<<<EWB, 256, 0, stream>>>(S2, crossNE, (int)TOT);
    k_ko<<<4, 256, 0, stream>>>(S2, ko_mask, 1024);

    // ---- internal gconv: m = cross_c @ ienc_W ; acc = m + scatter ; z = leaky(acc+b) + x_c
    gemm(stream, S2, 512, iencWt, 512, nullptr, S3, 512, NROW, 512, 512, 0, 0);
    hipMemcpyAsync(S4, S3, BIGF * sizeof(float), hipMemcpyDeviceToDevice, stream);
    k_edge<<<131072, 256, 0, stream>>>(iedge, 131072, S3, S4);
    k_finz<<<EWB, 256, 0, stream>>>(S2, S4, ienc_b, S0, (int)TOT);

    // ---- z += pre_c @ pre_W + pre_b ----
    k_pad511<<<EWB, 256, 0, stream>>>(S1, pre_x, (int)TOT);
    k_ko<<<4, 256, 0, stream>>>(S1, ko_mask, 1024);
    gemm(stream, S1, 512, preWt, 512, pre_b, S2, 512, NROW, 512, 512, 0, 1);

    // ---- main gconv: m2 = z @ enc_W ; acc2 ; z2 = leaky(acc2 + enc_b) ----
    gemm(stream, S2, 512, encWt, 512, nullptr, S3, 512, NROW, 512, 512, 0, 0);
    hipMemcpyAsync(S4, S3, BIGF * sizeof(float), hipMemcpyDeviceToDevice, stream);
    k_edge<<<262144, 256, 0, stream>>>(edge, 262144, S3, S4);
    k_finz<<<EWB, 256, 0, stream>>>(S1, S4, enc_b, nullptr, (int)TOT);

    // ---- readout ----
    k_gather<<<BK_, 256, 0, stream>>>(S0, S1, ko_mask);                       // zk
    gemm(stream, S0, 512, g1Wt, 512, gate_b1, S3, 512, BK_, 512, 512, 2, 0);  // g = tanh(...)
    float* scores = S4;
    float* tvals  = S4 + 1024;
    k_rowdot<<<BK_ / 8, 256, 0, stream>>>(S3, S0, gate_W2, gate_b2, reg_W, scores, tvals);
    k_softmax_out<<<16, 64, 0, stream>>>(scores, tvals, reg_b, out);
}
